// Attention_91293824844283
// MI455X (gfx1250) — matrix operand order repaired and checked by reading, not compiled, not run on hardware
//
#include <hip/hip_runtime.h>

// ---------- vector types ----------
typedef __attribute__((ext_vector_type(16))) _Float16 f16x16;
typedef __attribute__((ext_vector_type(8)))  _Float16 f16x8;
typedef __attribute__((ext_vector_type(8)))  float    f32x8;

#define B_  4
#define N_  2048
#define D_  512
#define H_  8
#define E_  64

__device__ __forceinline__ f32x8 wmma_f16(f16x16 a, f16x16 b, f32x8 c) {
    // v_wmma_f32_16x16x32_f16 : D = A(16x32 f16) * B(32x16 f16) + C(16x16 f32)
    return __builtin_amdgcn_wmma_f32_16x16x32_f16(false, a, false, b, (short)0, c, false, false);
}

// LDS byte offset = low 32 bits of the generic pointer (addrspacecast truncation)
__device__ __forceinline__ uint32_t lds_off(const void* p) {
    return (uint32_t)(uintptr_t)p;
}
// 16B per-lane async copy global -> LDS (ASYNCcnt-tracked, no VGPR data)
__device__ __forceinline__ void async_copy_b128(uint32_t lds, const _Float16* g) {
    asm volatile("global_load_async_to_lds_b128 %0, %1, off"
                 :: "v"(lds), "v"((unsigned long long)(uintptr_t)g)
                 : "memory");
}
__device__ __forceinline__ void wait_async_le2() { asm volatile("s_wait_asynccnt 0x2" ::: "memory"); }
__device__ __forceinline__ void wait_async_0()   { asm volatile("s_wait_asynccnt 0x0" ::: "memory"); }

// A operand (16x32 f16), hardware layout:
//  lane<16 : elements 0..7 -> K = k0+0..7,   elements 8..15 -> K = k0+16..23
//  lane>=16: elements 0..7 -> K = k0+8..15,  elements 8..15 -> K = k0+24..31
__device__ __forceinline__ f16x16 load_a_f16(const _Float16* base, int ld, int row, int k0, int hf) {
    const _Float16* p = base + (size_t)row * ld + k0 + 8 * hf;
    union { f16x16 v; f16x8 h[2]; } u;
    u.h[0] = *(const f16x8*)(p);
    u.h[1] = *(const f16x8*)(p + 16);
    return u.v;
}
__device__ __forceinline__ f16x16 load_a_f32cvt(const float* base, int ld, int row, int k0, int hf) {
    const float* p = base + (size_t)row * ld + k0 + 8 * hf;
    f16x16 r;
#pragma unroll
    for (int i = 0; i < 8; ++i) r[i] = (_Float16)p[i];
#pragma unroll
    for (int i = 0; i < 8; ++i) r[8 + i] = (_Float16)p[16 + i];
    return r;
}

// B operand (32x16 f16): lane holds column (lane&15), elements 0..7 -> K = k0 + 8*(lane>=16) + j, elements 8..15 -> K = k0 + 16 + 8*(lane>=16) + (j-8).
// base indexed [col][k] with stride ld.
__device__ __forceinline__ f16x16 load_b_f16(const _Float16* base, int ld, int col, int k0, int hf) {
    const _Float16* p = base + (size_t)col * ld + k0 + 8 * hf;
    union { f16x16 v; f16x8 h[2]; } u;
    u.h[0] = *(const f16x8*)(p);
    u.h[1] = *(const f16x8*)(p + 16);
    return u.v;
}

// ======================================================================
// Kernel 0a: elementwise f32 -> f16 (same layout); n8 = elems/8
// ======================================================================
__global__ __launch_bounds__(256) void cvt_f16_kernel(const float* __restrict__ src,
                                                      _Float16* __restrict__ dst, int n8) {
    const int i = blockIdx.x * 256 + threadIdx.x;
    if (i < n8) {
        const float* p = src + (size_t)i * 8;
        f16x8 r;
#pragma unroll
        for (int k = 0; k < 8; ++k) r[k] = (_Float16)p[k];
        *(f16x8*)(dst + (size_t)i * 8) = r;
    }
}

// ======================================================================
// Kernel 0b: Wo[h][o][e] (f32) -> WoT[o][h*64+e] (f16)
// 262144 elems, 8 per thread -> 32768 threads -> grid 128 x 256
// ======================================================================
__global__ __launch_bounds__(256) void cvt_woT_kernel(const float* __restrict__ Wo,
                                                      _Float16* __restrict__ WoT) {
    const int flat = blockIdx.x * 256 + threadIdx.x;   // 0..32767
    const int h = flat >> 12;                           // 4096 threads per head
    const int rem = flat & 4095;
    const int o = rem >> 3;
    const int e0 = (rem & 7) * 8;
    const float* p = Wo + ((size_t)h * D_ + o) * E_ + e0;
    f16x8 r;
#pragma unroll
    for (int k = 0; k < 8; ++k) r[k] = (_Float16)p[k];
    *(f16x8*)(WoT + (size_t)o * D_ + h * E_ + e0) = r;
}

// ======================================================================
// Kernel 1: per-head projection  Out[b,h,n,e] = sum_d X[b,n,d] * W16[h,e,d]
// grid = (N/128, B*H), block = 256. Whole 64x512 f16 weight head staged
// into LDS with async-to-LDS copies; B operands come from LDS.
// ======================================================================
__global__ __launch_bounds__(256) void proj_kernel(const float* __restrict__ X,
                                                   const _Float16* __restrict__ W16,
                                                   _Float16* __restrict__ Out) {
    __shared__ __align__(16) _Float16 Wlds[E_][D_];   // 64 KB
    const int bh = blockIdx.y, b = bh >> 3, h = bh & 7;
    const int tid = threadIdx.x, lane = tid & 31, wid = tid >> 5;
    const int hf = lane >> 4, ln = lane & 15;
    const float* Xb = X + (size_t)b * N_ * D_;
    const _Float16* Wh = W16 + (size_t)h * E_ * D_;
    const int m0 = blockIdx.x * 128 + wid * 16;

    {   // async stage: each thread copies 256B (16 x b128), total 64KB
        const uint32_t lbase = lds_off(&Wlds[0][0]) + (uint32_t)tid * 256;
        const _Float16* g = Wh + (size_t)tid * 128;
#pragma unroll
        for (int c = 0; c < 16; ++c) async_copy_b128(lbase + c * 16, g + c * 8);
    }
    wait_async_0();
    __syncthreads();

    f32x8 acc[4] = {};
    for (int k0 = 0; k0 < D_; k0 += 32) {
        f16x16 a = load_a_f32cvt(Xb, D_, m0 + ln, k0, hf);
#pragma unroll
        for (int t = 0; t < 4; ++t) {
            f16x16 bm = load_b_f16(&Wlds[0][0], D_, t * 16 + ln, k0, hf);
            acc[t] = wmma_f16(a, bm, acc[t]);
        }
    }
    _Float16* ob = Out + (size_t)bh * N_ * E_;
#pragma unroll
    for (int t = 0; t < 4; ++t)
#pragma unroll
        for (int r = 0; r < 8; ++r)
            ob[(size_t)(m0 + r + 8 * hf) * E_ + t * 16 + ln] = (_Float16)acc[t][r];
}

// ======================================================================
// Kernel 2: V transpose per (b,h):  Vt[e][n] = V[n][e]  (f16)
// ======================================================================
__global__ __launch_bounds__(256) void transpose_kernel(const _Float16* __restrict__ V,
                                                        _Float16* __restrict__ Vt) {
    __shared__ __align__(16) _Float16 t[64][72];   // pad: row stride 144B (16B multiple)
    const int bh = blockIdx.y;
    const int j0 = blockIdx.x * 64;
    const int tid = threadIdx.x;
    const _Float16* Vb = V + (size_t)bh * N_ * E_;
    _Float16* Vtb = Vt + (size_t)bh * E_ * N_;
    {
        const int j = tid >> 2, e0 = (tid & 3) * 16;
        *(f16x8*)&t[j][e0]     = *(const f16x8*)&Vb[(size_t)(j0 + j) * E_ + e0];
        *(f16x8*)&t[j][e0 + 8] = *(const f16x8*)&Vb[(size_t)(j0 + j) * E_ + e0 + 8];
    }
    __syncthreads();
    {
        const int e = tid >> 2, c0 = (tid & 3) * 16;
        f16x8 r0, r1;
#pragma unroll
        for (int i = 0; i < 8; ++i) r0[i] = t[c0 + i][e];
#pragma unroll
        for (int i = 0; i < 8; ++i) r1[i] = t[c0 + 8 + i][e];
        *(f16x8*)&Vtb[(size_t)e * N_ + j0 + c0]     = r0;
        *(f16x8*)&Vtb[(size_t)e * N_ + j0 + c0 + 8] = r1;
    }
}

// ======================================================================
// Kernel 3: flash attention per (b,h). grid = (N/128, B*H), block = 256.
// Double-buffered async-to-LDS K/Vt staging (ASYNCcnt), online softmax.
// ======================================================================
__global__ __launch_bounds__(256) void attn_kernel(const _Float16* __restrict__ Q,
                                                   const _Float16* __restrict__ K,
                                                   const _Float16* __restrict__ Vt,
                                                   _Float16* __restrict__ rep) {
    __shared__ __align__(16) _Float16 Klds[2][32][E_];    // [buf][j][e]
    __shared__ __align__(16) _Float16 Vtlds[2][E_][32];   // [buf][e][j]
    __shared__ __align__(16) _Float16 Plds[8][16][32];    // per-wave P staging

    const int bh = blockIdx.y;
    const int tid = threadIdx.x, lane = tid & 31, wid = tid >> 5;
    const int hf = lane >> 4, ln = lane & 15;
    const _Float16* Qb  = Q  + (size_t)bh * N_ * E_;
    const _Float16* Kb  = K  + (size_t)bh * N_ * E_;
    const _Float16* Vtb = Vt + (size_t)bh * E_ * N_;
    const int m0 = blockIdx.x * 128 + wid * 16;

    // Q operands (scale 1/sqrt(64)=0.125 folded in, exact in f16)
    f16x16 qa0 = load_a_f16(Qb, E_, m0 + ln, 0, hf);
    f16x16 qa1 = load_a_f16(Qb, E_, m0 + ln, 32, hf);
#pragma unroll
    for (int i = 0; i < 16; ++i) {
        qa0[i] = qa0[i] * (_Float16)0.125f;
        qa1[i] = qa1[i] * (_Float16)0.125f;
    }

    // per-thread async staging assignment (16B each, 4KB per tile)
    const int kj = tid >> 3, ke = (tid & 7) * 8;   // K:  [32][64]
    const int ve = tid >> 2, vj = (tid & 3) * 8;   // Vt: [64][32]
    const _Float16* Kg = Kb  + (size_t)kj * E_ + ke;
    const _Float16* Vg = Vtb + (size_t)ve * N_ + vj;
    const uint32_t kl[2] = { lds_off(&Klds[0][kj][ke]), lds_off(&Klds[1][kj][ke]) };
    const uint32_t vl[2] = { lds_off(&Vtlds[0][ve][vj]), lds_off(&Vtlds[1][ve][vj]) };

    f32x8 o[4] = {};
    float mrun[8], lrun[8];
#pragma unroll
    for (int r = 0; r < 8; ++r) { mrun[r] = -1e30f; lrun[r] = 0.0f; }
    const float L2E = 1.44269504088896f;

    // prologue: stage tile 0
    async_copy_b128(kl[0], Kg);
    async_copy_b128(vl[0], Vg);

    int buf = 0;
    for (int j0 = 0; j0 < N_; j0 += 32) {
        const bool more = (j0 + 32) < N_;
        if (more) {   // prefetch next tile into the other buffer
            async_copy_b128(kl[buf ^ 1], Kg + (size_t)(j0 + 32) * E_);
            async_copy_b128(vl[buf ^ 1], Vg + (j0 + 32));
        }
        if (more) wait_async_le2(); else wait_async_0();   // current tile complete (this wave)
        __syncthreads();                                    // ... and for every wave

        const _Float16* Kt  = &Klds[buf][0][0];
        const _Float16* Vb2 = &Vtlds[buf][0][0];

        // S = (Q/8) K^T, two 16-wide column tiles, k over E=64 -> 2 WMMAs each
        f32x8 s0 = {}, s1 = {};
        s0 = wmma_f16(qa0, load_b_f16(Kt, E_, ln,      0,  hf), s0);
        s0 = wmma_f16(qa1, load_b_f16(Kt, E_, ln,      32, hf), s0);
        s1 = wmma_f16(qa0, load_b_f16(Kt, E_, 16 + ln, 0,  hf), s1);
        s1 = wmma_f16(qa1, load_b_f16(Kt, E_, 16 + ln, 32, hf), s1);

        // online softmax; C-layout: VGPR r, lanes<16 -> row r, lanes>=16 -> row r+8
        float corr[8];
#pragma unroll
        for (int r = 0; r < 8; ++r) {
            float a = s0[r], c2 = s1[r];
            float mx = fmaxf(a, c2);
            mx = fmaxf(mx, __shfl_xor(mx, 1));
            mx = fmaxf(mx, __shfl_xor(mx, 2));
            mx = fmaxf(mx, __shfl_xor(mx, 4));
            mx = fmaxf(mx, __shfl_xor(mx, 8));
            float mn = fmaxf(mrun[r], mx);
            float c  = exp2f((mrun[r] - mn) * L2E);
            float p0 = exp2f((a  - mn) * L2E);
            float p1 = exp2f((c2 - mn) * L2E);
            float rs = p0 + p1;
            rs += __shfl_xor(rs, 1);
            rs += __shfl_xor(rs, 2);
            rs += __shfl_xor(rs, 4);
            rs += __shfl_xor(rs, 8);
            lrun[r] = lrun[r] * c + rs;
            mrun[r] = mn;
            corr[r] = c;
            Plds[wid][r + 8 * hf][ln]      = (_Float16)p0;
            Plds[wid][r + 8 * hf][16 + ln] = (_Float16)p1;
        }
#pragma unroll
        for (int t = 0; t < 4; ++t)
#pragma unroll
            for (int r = 0; r < 8; ++r) o[t][r] *= corr[r];

        // P is wave-private; HW keeps same-wave DS ops in order — only stop
        // the compiler from reordering the re-layout read above the stores.
        asm volatile("" ::: "memory");

        // O += P V : A = P (16x32 via LDS re-layout), B = V^T tiles
        f16x16 pa = load_a_f16(&Plds[wid][0][0], 32, ln, 0, hf);
#pragma unroll
        for (int t = 0; t < 4; ++t)
            o[t] = wmma_f16(pa, load_b_f16(Vb2, 32, t * 16 + ln, 0, hf), o[t]);

        __syncthreads();   // all reads of `buf` done -> safe target for next prefetch
        buf ^= 1;
    }

    // epilogue: normalize, write rep[b, i, h*64 + e] (f16, row stride 512)
    const int b = bh >> 3, h = bh & 7;
    _Float16* rb = rep + (size_t)b * N_ * D_ + h * E_;
#pragma unroll
    for (int r = 0; r < 8; ++r) {
        float inv = 1.0f / lrun[r];
#pragma unroll
        for (int t = 0; t < 4; ++t)
            rb[(size_t)(m0 + r + 8 * hf) * D_ + t * 16 + ln] = (_Float16)(o[t][r] * inv);
    }
}

// ======================================================================
// Kernel 4: out = R[8192 x 512] @ WoT[512 x 512] (f16 x f16 -> f32)
// grid = (512/128, 8192/128), block = 256
// ======================================================================
__global__ __launch_bounds__(256) void outproj_kernel(const _Float16* __restrict__ R,
                                                      const _Float16* __restrict__ WoT,
                                                      float* __restrict__ Out) {
    const int tid = threadIdx.x, lane = tid & 31, wid = tid >> 5;
    const int hf = lane >> 4, ln = lane & 15;
    const int row0 = blockIdx.y * 128 + wid * 16;   // over B*N
    const int c0 = blockIdx.x * 128;

    f32x8 acc[8] = {};
    for (int k0 = 0; k0 < D_; k0 += 32) {
        f16x16 a = load_a_f16(R, D_, row0 + ln, k0, hf);
#pragma unroll
        for (int t = 0; t < 8; ++t) {
            f16x16 bm = load_b_f16(WoT, D_, c0 + t * 16 + ln, k0, hf);
            acc[t] = wmma_f16(a, bm, acc[t]);
        }
    }
#pragma unroll
    for (int t = 0; t < 8; ++t)
#pragma unroll
        for (int r = 0; r < 8; ++r)
            Out[(size_t)(row0 + r + 8 * hf) * D_ + c0 + t * 16 + ln] = acc[t][r];
}

// ======================================================================
extern "C" void kernel_launch(void* const* d_in, const int* in_sizes, int n_in,
                              void* d_out, int out_size, void* d_ws, size_t ws_size,
                              hipStream_t stream) {
    (void)in_sizes; (void)n_in; (void)out_size; (void)ws_size;
    const float* x1 = (const float*)d_in[0];
    const float* x2 = (const float*)d_in[1];
    const float* v  = (const float*)d_in[2];
    const float* Wq = (const float*)d_in[3];
    const float* Wk = (const float*)d_in[4];
    const float* Wv = (const float*)d_in[5];
    const float* Wo = (const float*)d_in[6];
    float* out = (float*)d_out;

    const size_t qkvElems = (size_t)B_ * H_ * N_ * E_;   // == B*N*D
    const size_t wElems   = (size_t)H_ * E_ * D_;        // 262144
    _Float16* Qw  = (_Float16*)d_ws;
    _Float16* Kw  = Qw + qkvElems;
    _Float16* Vtw = Kw + qkvElems;
    _Float16* Vw  = Vtw + qkvElems;      // dead after transpose -> aliased by Rw
    _Float16* Rw  = Vw;
    _Float16* Wqh = Vw + qkvElems;
    _Float16* Wkh = Wqh + wElems;
    _Float16* Wvh = Wkh + wElems;
    _Float16* WoT = Wvh + wElems;

    dim3 blk(256);

    // weights -> f16 (once)
    const int w8 = (int)(wElems / 8);                    // 32768
    cvt_f16_kernel<<<dim3(w8 / 256), blk, 0, stream>>>(Wq, Wqh, w8);
    cvt_f16_kernel<<<dim3(w8 / 256), blk, 0, stream>>>(Wk, Wkh, w8);
    cvt_f16_kernel<<<dim3(w8 / 256), blk, 0, stream>>>(Wv, Wvh, w8);
    cvt_woT_kernel<<<dim3(w8 / 256), blk, 0, stream>>>(Wo, WoT);

    dim3 gProj(N_ / 128, B_ * H_);
    proj_kernel<<<gProj, blk, 0, stream>>>(x2, Wqh, Qw);   // queries
    proj_kernel<<<gProj, blk, 0, stream>>>(x1, Wkh, Kw);   // keys
    proj_kernel<<<gProj, blk, 0, stream>>>(v,  Wvh, Vw);   // values

    transpose_kernel<<<dim3(N_ / 64, B_ * H_), blk, 0, stream>>>(Vw, Vtw);

    attn_kernel<<<dim3(N_ / 128, B_ * H_), blk, 0, stream>>>(Qw, Kw, Vtw, Rw);

    outproj_kernel<<<dim3(D_ / 128, (B_ * N_) / 128), blk, 0, stream>>>(Rw, WoT, out);
}
